// STGCN_Block_69011534512917
// MI455X (gfx1250) — compile-verified, matrix operand order checked by reading, not run on hardware
//
#include <hip/hip_runtime.h>
#include <hip/hip_bf16.h>

// ---------------- problem constants ----------------
constexpr int kN    = 16;
constexpr int kCIN  = 128;
constexpr int kT    = 1024;
constexpr int kV    = 25;
constexpr int kVP   = 32;    // v padded to 32 for WMMA K
constexpr int kK3   = 3;
constexpr int kCOUT = 128;
constexpr int kTK   = 9;
constexpr int kPAD  = 4;
constexpr float kEPS = 1e-5f;

// ---------------- WMMA fragment types ----------------
typedef __bf16 bf16x16 __attribute__((ext_vector_type(16)));
typedef float  f32x8   __attribute__((ext_vector_type(8)));
typedef unsigned short u16x8 __attribute__((ext_vector_type(8)));

union Frag {
  bf16x16 bf;
  u16x8   h[2];
  unsigned short u[16];
};

__device__ __forceinline__ unsigned short f2bf(float f) {
  unsigned u = __builtin_bit_cast(unsigned, f);
  u += 0x7fffu + ((u >> 16) & 1u);   // round-to-nearest-even
  return (unsigned short)(u >> 16);
}

#define WMMA_BF16(A, B, C) __builtin_amdgcn_wmma_f32_16x16x32_bf16( \
    false, (A), false, (B), (short)0, (C), false, false)

// ---------------- 1) InstanceNorm statistics ----------------
__global__ void __launch_bounds__(256) stats_kernel(
    const float* __restrict__ x, float* __restrict__ scale, float* __restrict__ shift) {
  __shared__ float ss[256];
  __shared__ float sq[256];
  const int nc = blockIdx.x;                       // (n*128 + c)
  const float* p = x + (size_t)nc * (kT * kV);
  float s = 0.f, q = 0.f;
  for (int i = threadIdx.x; i < kT * kV; i += 256) {
    float v = p[i];
    s += v; q += v * v;
  }
  ss[threadIdx.x] = s; sq[threadIdx.x] = q;
  __syncthreads();
  for (int off = 128; off > 0; off >>= 1) {
    if ((int)threadIdx.x < off) {
      ss[threadIdx.x] += ss[threadIdx.x + off];
      sq[threadIdx.x] += sq[threadIdx.x + off];
    }
    __syncthreads();
  }
  if (threadIdx.x == 0) {
    const float inv = 1.f / (float)(kT * kV);
    float mu  = ss[0] * inv;
    float var = sq[0] * inv - mu * mu;
    float sc  = rsqrtf(var + kEPS);
    scale[nc] = sc;
    shift[nc] = -mu * sc;
  }
}

// ---------------- 2) normalize + ReLU -> bf16, transposed to [n][t][v32][ci] ----
__global__ void __launch_bounds__(256) norm_kernel(
    const float* __restrict__ x, const float* __restrict__ scale,
    const float* __restrict__ shift, unsigned short* __restrict__ xn2) {
  __shared__ __align__(16) unsigned short lds[400 * 64];   // (16t*25v) x 64c, 50 KB
  const int t0 = blockIdx.x * 16;
  const int nn = blockIdx.y;
  const int cb = blockIdx.z * 64;                          // channel half
  #pragma unroll
  for (int i = 0; i < 100; ++i) {
    int flat = threadIdx.x + i * 256;                      // 64c * 400tv
    int cl = flat / 400, tv = flat % 400;
    int nc = nn * kCIN + cb + cl;
    float val = x[((size_t)nc * kT + t0) * kV + tv] * scale[nc] + shift[nc];
    val = val > 0.f ? val : 0.f;
    lds[tv * 64 + cl] = f2bf(val);
  }
  __syncthreads();
  #pragma unroll
  for (int i = 0; i < 16; ++i) {
    int chunk = threadIdx.x + i * 256;                     // 4096 chunks of 8
    int tv32 = chunk >> 3;
    int clb = (chunk & 7) << 3;
    int t = tv32 >> 5, v = tv32 & 31;
    u16x8 valv = {};
    if (v < kV) valv = *(const u16x8*)&lds[(t * kV + v) * 64 + clb];
    *(u16x8*)&xn2[(((size_t)nn * kT + t0 + t) * kVP + v) * kCIN + cb + clb] = valv;
  }
}

// ---------------- 3) weight prep ----------------
__global__ void __launch_bounds__(256) prep_kernel(
    const float* __restrict__ A, const float* __restrict__ wgcn,
    const float* __restrict__ wtcn,
    unsigned short* __restrict__ wg, unsigned short* __restrict__ apf,
    unsigned short* __restrict__ wt2) {
  const int idx = blockIdx.x * 256 + threadIdx.x;
  if (idx < kK3 * kCOUT * kCIN) wg[idx] = f2bf(wgcn[idx]);
  if (idx < kK3 * 2 * 32 * 16) {
    int k = idx / 1024, r = idx % 1024;
    int wt = r >> 9, rr = r & 511;
    int lane = rr >> 4, e = rr & 15;
    int koff = (lane >> 4) * 8, l15 = lane & 15;
    int v = (e < 8) ? (koff + e) : (koff + 8 + e);     // ISA bf16 K-chunk order
    int w = wt * 16 + l15;
    apf[idx] = f2bf((v < kV && w < kV) ? A[(k * kV + v) * kV + w] : 0.f);
  }
  if (idx < kCOUT * kTK * kCOUT) {
    int co = idx / (kTK * kCOUT), r = idx % (kTK * kCOUT);
    int dt = r / kCOUT, ci = r % kCOUT;
    wt2[idx] = f2bf(wtcn[((size_t)co * kCOUT + ci) * kTK + dt]);  // w_tcn [co][ci][9][1]
  }
}

// ---------------- 4) fused 1x1-conv GEMM + graph contraction (WMMA) ----------------
__global__ void __launch_bounds__(256) gcn_kernel(
    const unsigned short* __restrict__ xn2,
    const unsigned short* __restrict__ wg,
    const unsigned short* __restrict__ apf,
    const float* __restrict__ bgcn,
    unsigned short* __restrict__ y2) {
  __shared__ __align__(16) unsigned short ldsY[512 * 32];  // rows=(t*32+c), cols=v32
  const int t0 = blockIdx.x * 16;
  const int nn = blockIdx.y;
  const int ch = blockIdx.z;                 // output channel quarter (32 channels)
  const int tid = threadIdx.x;
  const int lane = tid & 31, wv = tid >> 5;
  const int l15 = lane & 15, hi = lane >> 4;
  const int koff = hi * 8;                   // per-lane K-chunk base (ISA bf16 layout)

  f32x8 y2acc[4][2] = {};                    // 4 row-tiles x 2 w-tiles per wave

  for (int k = 0; k < kK3; ++k) {
    f32x8 g1[2][4] = {};                     // GEMM1: 2 M-tiles x 4 col-tiles per wave
    for (int ci0 = 0; ci0 < kCIN; ci0 += 32) {
      Frag a0, a1;                           // weights, K=ci contiguous
      {
        int o = k * kCOUT + ch * 32 + l15;
        const unsigned short* w0 = wg + (size_t)o * kCIN + ci0 + koff;
        a0.h[0] = *(const u16x8*)w0;  a0.h[1] = *(const u16x8*)(w0 + 16);
        const unsigned short* w1 = wg + (size_t)(o + 16) * kCIN + ci0 + koff;
        a1.h[0] = *(const u16x8*)w1;  a1.h[1] = *(const u16x8*)(w1 + 16);
      }
      #pragma unroll
      for (int ctl = 0; ctl < 4; ++ctl) {
        int ct = wv * 4 + ctl;
        int tl = ct >> 1;                    // t_local uniform per tile
        int v = (ct & 1) * 16 + l15;         // includes zero-padded v>=25
        const unsigned short* bp =
            xn2 + (((size_t)nn * kT + t0 + tl) * kVP + v) * kCIN + ci0 + koff;
        Frag b;
        b.h[0] = *(const u16x8*)bp;  b.h[1] = *(const u16x8*)(bp + 16);
        g1[0][ctl] = WMMA_BF16(a0.bf, b.bf, g1[0][ctl]);
        g1[1][ctl] = WMMA_BF16(a1.bf, b.bf, g1[1][ctl]);
      }
    }
    // spill y1 (+bias) to LDS as bf16, rows = (t_local*32 + c), cols = v32.
    for (int mt = 0; mt < 2; ++mt)
      for (int ctl = 0; ctl < 4; ++ctl) {
        int ct = wv * 4 + ctl;
        int tl = ct >> 1, v = (ct & 1) * 16 + l15;
        #pragma unroll
        for (int r = 0; r < 8; ++r) {
          int c2 = mt * 16 + r + hi * 8;
          float val = g1[mt][ctl][r] + bgcn[k * kCOUT + ch * 32 + c2];
          ldsY[(tl * 32 + c2) * 32 + v] = f2bf(val);
        }
      }
    __syncthreads();
    // graph contraction: one K=32 WMMA per (row-tile, w-tile)
    #pragma unroll
    for (int rtl = 0; rtl < 4; ++rtl) {
      int rt = wv * 4 + rtl;
      int row = rt * 16 + l15;
      Frag a;
      a.h[0] = *(const u16x8*)&ldsY[row * 32 + koff];
      a.h[1] = *(const u16x8*)&ldsY[row * 32 + koff + 16];
      #pragma unroll
      for (int wt = 0; wt < 2; ++wt) {
        Frag b;                              // pre-swizzled fragment: 2 x b128
        const unsigned short* ap = apf + (((k * 2 + wt) * 32) + lane) * 16;
        b.h[0] = *(const u16x8*)ap;  b.h[1] = *(const u16x8*)(ap + 8);
        y2acc[rtl][wt] = WMMA_BF16(a.bf, b.bf, y2acc[rtl][wt]);
      }
    }
    __syncthreads();
  }
  // epilogue: rows are (t, c) with c along VGPR index -> packed b128 stores
  for (int rtl = 0; rtl < 4; ++rtl) {
    int rt = wv * 4 + rtl;
    int t = rt >> 1, cbase = (rt & 1) * 16;
    for (int wt = 0; wt < 2; ++wt) {
      int w = wt * 16 + l15;
      if (w < kV) {
        Frag pk;
        #pragma unroll
        for (int r = 0; r < 8; ++r) pk.u[r] = f2bf(y2acc[rtl][wt][r]);
        *(u16x8*)&y2[(((size_t)nn * kT + t0 + t) * kV + w) * kCOUT +
                     ch * 32 + cbase + hi * 8] = pk.h[0];
      }
    }
  }
}

// ---------------- 5) temporal conv (9,1) with reflect pad, WMMA ----------------
// grid (64 t-tiles, 16 n, 2 co-halves). K = (dt, ci) = 1152 -> 36 WMMA/tile.
// Interior t-tiles (62 of 64) take a fast path: per-column base pointers are
// hoisted and every fragment load folds into a base + immediate-offset b128.
__global__ void __launch_bounds__(256) tcn_kernel(
    const unsigned short* __restrict__ y2,   // [n][t][25][128] bf16
    const unsigned short* __restrict__ wt2,  // [co][9][128]    bf16
    const float* __restrict__ btcn,
    float* __restrict__ out) {               // [n][co][t][25]  fp32
  const int t0 = blockIdx.x * 16;
  const int nn = blockIdx.y;
  const int cob = blockIdx.z * 64;
  const int tid = threadIdx.x;
  const int lane = tid & 31, wv = tid >> 5;
  const int l15 = lane & 15, hi = lane >> 4;
  const int koff = hi * 8;
  const int mtw = (wv & 1) * 2;              // 2 of 4 co-tiles
  const int ctw = (wv >> 1) * 8;             // 8 of 32 (t,w) col-tiles

  f32x8 acc[2][8] = {};
  // weight base pointers: [co][dt][ci] -> inner offsets dt*128 + ci0 are immediates
  const unsigned short* wa0 =
      wt2 + (size_t)(cob + mtw * 16 + l15) * (kTK * kCOUT) + koff;
  const unsigned short* wa1 = wa0 + 16 * (kTK * kCOUT);

  const bool interior = (t0 >= kPAD) && (t0 + 15 + kPAD <= kT - 1);
  if (interior) {
    // per-column base pointers at dt=0 (t offset -kPAD); dt step = kV*kCOUT elems
    const unsigned short* pb[8];
    #pragma unroll
    for (int c8 = 0; c8 < 8; ++c8) {
      int ct = ctw + c8;
      int tl = ct >> 1;
      int w = (ct & 1) * 16 + l15; if (w > 24) w = 24;
      pb[c8] = y2 + (((size_t)nn * kT + (t0 + tl - kPAD)) * kV + w) * kCOUT + koff;
    }
    for (int dt = 0; dt < kTK; ++dt) {
      #pragma unroll
      for (int cc = 0; cc < 4; ++cc) {
        const int ci0 = cc * 32;
        Frag a0, a1;
        a0.h[0] = *(const u16x8*)(wa0 + ci0);       a0.h[1] = *(const u16x8*)(wa0 + ci0 + 16);
        a1.h[0] = *(const u16x8*)(wa1 + ci0);       a1.h[1] = *(const u16x8*)(wa1 + ci0 + 16);
        #pragma unroll
        for (int c8 = 0; c8 < 8; ++c8) {
          Frag b;
          b.h[0] = *(const u16x8*)(pb[c8] + ci0);
          b.h[1] = *(const u16x8*)(pb[c8] + ci0 + 16);
          acc[0][c8] = WMMA_BF16(a0.bf, b.bf, acc[0][c8]);
          acc[1][c8] = WMMA_BF16(a1.bf, b.bf, acc[1][c8]);
        }
      }
      // advance bases to next tap (constant pointer bumps, no per-load math)
      wa0 += kCOUT; wa1 += kCOUT;
      #pragma unroll
      for (int c8 = 0; c8 < 8; ++c8) pb[c8] += kV * kCOUT;
    }
  } else {
    // border tiles: reflect padding on t
    for (int dt = 0; dt < kTK; ++dt) {
      for (int ci0 = 0; ci0 < kCOUT; ci0 += 32) {
        Frag a0, a1;
        a0.h[0] = *(const u16x8*)(wa0 + dt * kCOUT + ci0);
        a0.h[1] = *(const u16x8*)(wa0 + dt * kCOUT + ci0 + 16);
        a1.h[0] = *(const u16x8*)(wa1 + dt * kCOUT + ci0);
        a1.h[1] = *(const u16x8*)(wa1 + dt * kCOUT + ci0 + 16);
        for (int c8 = 0; c8 < 8; ++c8) {
          int ct = ctw + c8;
          int tl = ct >> 1;
          int w = (ct & 1) * 16 + l15; if (w > 24) w = 24;
          int tr = t0 + tl + dt - kPAD;
          tr = tr < 0 ? -tr : tr;
          tr = tr > kT - 1 ? 2 * (kT - 1) - tr : tr;
          const unsigned short* bp =
              y2 + (((size_t)nn * kT + tr) * kV + w) * kCOUT + ci0 + koff;
          Frag b;
          b.h[0] = *(const u16x8*)bp;
          b.h[1] = *(const u16x8*)(bp + 16);
          acc[0][c8] = WMMA_BF16(a0.bf, b.bf, acc[0][c8]);
          acc[1][c8] = WMMA_BF16(a1.bf, b.bf, acc[1][c8]);
        }
      }
    }
  }
  for (int i = 0; i < 2; ++i)
    for (int c8 = 0; c8 < 8; ++c8) {
      int ct = ctw + c8;
      int tl = ct >> 1;
      int w = (ct & 1) * 16 + l15;
      if (w < kV) {
        #pragma unroll
        for (int r = 0; r < 8; ++r) {
          int co = cob + (mtw + i) * 16 + r + hi * 8;
          out[(((size_t)nn * kCOUT + co) * kT + (t0 + tl)) * kV + w] = acc[i][c8][r] + btcn[co];
        }
      }
    }
}

// ---------------- launcher ----------------
extern "C" void kernel_launch(void* const* d_in, const int* in_sizes, int n_in,
                              void* d_out, int out_size, void* d_ws, size_t ws_size,
                              hipStream_t stream) {
  (void)in_sizes; (void)n_in; (void)out_size; (void)ws_size;
  const float* x    = (const float*)d_in[0];
  const float* A    = (const float*)d_in[1];
  const float* wgcn = (const float*)d_in[2];
  const float* bgcn = (const float*)d_in[3];
  const float* wtcn = (const float*)d_in[4];
  const float* btcn = (const float*)d_in[5];
  float* out = (float*)d_out;

  char* ws = (char*)d_ws;
  float* scale          = (float*)(ws);                                 //   8 KB
  float* shift          = (float*)(ws + 8192);                          //   8 KB
  unsigned short* wg    = (unsigned short*)(ws + 16384);                //  96 KB
  unsigned short* apf   = (unsigned short*)(ws + 16384 + 98304);        //   6 KB
  unsigned short* wt2   = (unsigned short*)(ws + 16384 + 98304 + 6144); // 288 KB
  unsigned short* xn2   = (unsigned short*)(ws + 512 * 1024);           // 128 MiB
  unsigned short* y2    = (unsigned short*)(ws + 512 * 1024 +
                           (size_t)kN * kT * kVP * kCIN * 2);           // 100 MiB

  stats_kernel<<<kN * kCIN, 256, 0, stream>>>(x, scale, shift);
  norm_kernel<<<dim3(kT / 16, kN, 2), 256, 0, stream>>>(x, scale, shift, xn2);
  prep_kernel<<<(kCOUT * kTK * kCOUT + 255) / 256, 256, 0, stream>>>(
      A, wgcn, wtcn, wg, apf, wt2);
  gcn_kernel<<<dim3(kT / 16, kN, 4), 256, 0, stream>>>(xn2, wg, apf, bgcn, y2);
  tcn_kernel<<<dim3(kT / 16, kN, 2), 256, 0, stream>>>(y2, wt2, btcn, out);
}
